// SimpleNetwork_9079560864496
// MI455X (gfx1250) — matrix operand order repaired and checked by reading, not compiled, not run on hardware
//
#include <hip/hip_runtime.h>

// CDNA5 / gfx1250: wave32, WMMA 16x16x32 f16 -> f32
typedef __attribute__((ext_vector_type(16))) _Float16 v16h;
typedef __attribute__((ext_vector_type(8)))  float    v8f;

#define TPB 256

// ---------------------------------------------------------------------------
// Edge MLP: ew[e] = (relu(attr @ W1 + b1) @ W2 + b2) ; also deg[dst] += ew
// ---------------------------------------------------------------------------
__global__ void edge_mlp_kernel(const float* __restrict__ attr,
                                const float* __restrict__ W1,
                                const float* __restrict__ b1,
                                const float* __restrict__ W2,
                                const float* __restrict__ b2,
                                const int*   __restrict__ dst,
                                float* __restrict__ ew,
                                float* __restrict__ deg,
                                int E)
{
    __shared__ float sW1[8 * 64];
    __shared__ float sb1[64];
    __shared__ float sW2[64];

    const int tid = threadIdx.x;
    for (int i = tid; i < 8 * 64; i += blockDim.x) sW1[i] = W1[i];
    if (tid < 64) { sb1[tid] = b1[tid]; sW2[tid] = W2[tid]; }
    __syncthreads();

    const int e = blockIdx.x * blockDim.x + tid;
    if (e >= E) return;

    float a[8];
#pragma unroll
    for (int i = 0; i < 8; ++i) a[i] = attr[(size_t)e * 8 + i];

    float w = b2[0];
    for (int j = 0; j < 64; ++j) {
        float s = sb1[j];
#pragma unroll
        for (int i = 0; i < 8; ++i) s = fmaf(a[i], sW1[i * 64 + j], s);
        s = fmaxf(s, 0.0f);
        w = fmaf(s, sW2[j], w);
    }
    ew[e] = w;
    atomicAdd(&deg[dst[e]], w);
}

__global__ void dinv_kernel(const float* __restrict__ deg,
                            float* __restrict__ dinv, int N)
{
    const int i = blockIdx.x * blockDim.x + threadIdx.x;
    if (i >= N) return;
    const float d = deg[i];
    dinv[i] = (d > 0.0f) ? rsqrtf(d) : 0.0f;
}

__global__ void norm_kernel(const float* __restrict__ ew,
                            const int*   __restrict__ src,
                            const int*   __restrict__ dst,
                            const float* __restrict__ dinv,
                            float* __restrict__ nrm, int E)
{
    const int e = blockIdx.x * blockDim.x + threadIdx.x;
    if (e >= E) return;
    nrm[e] = dinv[src[e]] * ew[e] * dinv[dst[e]];
}

// ---------------------------------------------------------------------------
// copy input features into stack columns [0, Hin) of node-major stack
// stk[n][Ktot] ; in[n][Hin]
// ---------------------------------------------------------------------------
__global__ void copy_cols_kernel(const float* __restrict__ in,
                                 float* __restrict__ stk,
                                 long total, int HinMask, int logHin, int Ktot)
{
    const long idx = (long)blockIdx.x * blockDim.x + threadIdx.x;
    if (idx >= total) return;
    const long n = idx >> logHin;
    const int  f = (int)(idx & (long)HinMask);
    stk[n * (size_t)Ktot + f] = in[idx];
}

// ---------------------------------------------------------------------------
// Propagation, H=64 features: one wave32 per edge, lane moves a float2.
// src/dst/nrm are wave-uniform -> scalar loads. Dominant HBM/atomic phase.
// hin/hout are stack pointers pre-offset to the right column block;
// stride = Ktot floats between node rows.
// ---------------------------------------------------------------------------
__global__ void propagate_w64(const float* __restrict__ hin,
                              float* __restrict__ hout,
                              const int*   __restrict__ src,
                              const int*   __restrict__ dst,
                              const float* __restrict__ nrm,
                              int E, int stride)
{
    const long wv   = ((long)blockIdx.x * blockDim.x + threadIdx.x) >> 5;
    const int  lane = threadIdx.x & 31;
    if (wv >= E) return;
    const int   s = src[wv];
    const int   d = dst[wv];
    const float w = nrm[wv];
    const float2 v = *(const float2*)(hin + (size_t)s * stride + 2 * lane);
    float* o = hout + (size_t)d * stride + 2 * lane;
    atomicAdd(o,     w * v.x);
    atomicAdd(o + 1, w * v.y);
}

// Propagation, H=16 features: one wave32 handles two edges (16 lanes each).
__global__ void propagate_w16(const float* __restrict__ hin,
                              float* __restrict__ hout,
                              const int*   __restrict__ src,
                              const int*   __restrict__ dst,
                              const float* __restrict__ nrm,
                              int E, int stride)
{
    const long wv   = ((long)blockIdx.x * blockDim.x + threadIdx.x) >> 5;
    const int  lane = threadIdx.x & 31;
    const long e    = wv * 2 + (lane >> 4);
    const int  f    = lane & 15;
    if (e >= E) return;
    const int   s = src[e];
    const int   d = dst[e];
    const float w = nrm[e];
    atomicAdd(hout + (size_t)d * stride + f, w * hin[(size_t)s * stride + f]);
}

// ---------------------------------------------------------------------------
// Pack f32 weights [Ktot][Hout] into f16 B fragments in CDNA5 WMMA layout:
//   frag[(c*NT + t)*32 + lane][16] halves, contiguous 32B per lane.
//   lanes 0-15:  N = t*16 + lane,    elems 0-7 -> K = c*32 + 0..7, elems 8-15 -> c*32 + 16..23
//   lanes 16-31: N = t*16 + lane-16, elems 0-7 -> K = c*32 + 8..15, elems 8-15 -> c*32 + 24..31
// ---------------------------------------------------------------------------
__global__ void pack_weights_kernel(const float* __restrict__ W,
                                    _Float16* __restrict__ frag,
                                    int Ktot, int Hout, int NT)
{
    const int idx = blockIdx.x * blockDim.x + threadIdx.x; // (c*NT+t)*32+lane
    const int nfrag = (Ktot / 32) * NT * 32;
    if (idx >= nfrag) return;
    const int lane = idx & 31;
    const int ct   = idx >> 5;
    const int t    = ct % NT;
    const int c    = ct / NT;
    const int col  = t * 16 + (lane & 15);
    const int kb   = c * 32 + ((lane >> 4) << 3);
    _Float16* o = frag + (size_t)idx * 16;
#pragma unroll
    for (int e = 0; e < 16; ++e) {
        const float v = (col < Hout) ? W[(size_t)(kb + (e & 7) + ((e >> 3) << 4)) * Hout + col] : 0.0f;
        o[e] = (_Float16)v;
    }
}

// ---------------------------------------------------------------------------
// Dense GEMM with WMMA: out[N,HOUT] = opt_relu(A[N,KTOT] @ W + bias)
// One wave per 16-row tile; NT = HOUT/16 column tiles share the A fragment.
// KTOT/32 fully-unrolled K chunks -> NCHUNK*NT v_wmma_f32_16x16x32_f16.
// A: 4x global_load_b128 + cvt per chunk; B: prepacked f16, one v16h per wmma.
// ---------------------------------------------------------------------------
template <int KTOT, int HOUT>
__global__ void gemm_wmma(const float* __restrict__ A,
                          const _Float16* __restrict__ Bfrag,
                          const float* __restrict__ bias,
                          float* __restrict__ out,
                          int nNodes, int doRelu)
{
    constexpr int NCHUNK = KTOT / 32;
    constexpr int NT     = (HOUT + 15) / 16;

    const int lane = threadIdx.x & 31;
    const int wave = threadIdx.x >> 5;
    const int mt   = blockIdx.x * (blockDim.x >> 5) + wave;
    const int tilesM = (nNodes + 15) >> 4;
    if (mt >= tilesM) return;          // whole-wave uniform exit
    const int m0 = mt << 4;

    // A fragment: lanes 0-15 M=lane (K blocks 0-7,16-23);
    //             lanes 16-31 M=lane-16 (K blocks 8-15,24-31)
    int arow = m0 + (lane & 15);
    if (arow >= nNodes) arow = nNodes - 1;       // clamp (N%16==0 normally)
    const int akb = (lane >> 4) << 3;            // 0 or 8
    const float* aptr = A + (size_t)arow * KTOT + akb;

    v8f acc[NT];
#pragma unroll
    for (int t = 0; t < NT; ++t) acc[t] = (v8f){};

#pragma unroll
    for (int c = 0; c < NCHUNK; ++c) {
        const float4 f0 = *(const float4*)(aptr + c * 32);
        const float4 f1 = *(const float4*)(aptr + c * 32 + 4);
        const float4 f2 = *(const float4*)(aptr + c * 32 + 16);
        const float4 f3 = *(const float4*)(aptr + c * 32 + 20);
        if (c + 1 < NCHUNK)
            __builtin_prefetch(aptr + (c + 1) * 32, 0, 3);  // global_prefetch_b8
        v16h Af;
        Af[0]  = (_Float16)f0.x;  Af[1]  = (_Float16)f0.y;
        Af[2]  = (_Float16)f0.z;  Af[3]  = (_Float16)f0.w;
        Af[4]  = (_Float16)f1.x;  Af[5]  = (_Float16)f1.y;
        Af[6]  = (_Float16)f1.z;  Af[7]  = (_Float16)f1.w;
        Af[8]  = (_Float16)f2.x;  Af[9]  = (_Float16)f2.y;
        Af[10] = (_Float16)f2.z;  Af[11] = (_Float16)f2.w;
        Af[12] = (_Float16)f3.x;  Af[13] = (_Float16)f3.y;
        Af[14] = (_Float16)f3.z;  Af[15] = (_Float16)f3.w;
#pragma unroll
        for (int t = 0; t < NT; ++t) {
            const v16h Bf = *(const v16h*)(Bfrag +
                (((size_t)c * NT + t) * 32 + lane) * 16);
            acc[t] = __builtin_amdgcn_wmma_f32_16x16x32_f16(
                false, Af, false, Bf, (short)0, acc[t], false, false);
        }
    }

    // C/D layout: VGPR v -> M = v (lanes 0-15) / v+8 (lanes 16-31); N = lane&15
    const int rbase = m0 + ((lane >> 4) << 3);
#pragma unroll
    for (int t = 0; t < NT; ++t) {
        const int col = t * 16 + (lane & 15);
        if (col >= HOUT) continue;                 // compile-time for HOUT=64
        const float bv = bias[col];
#pragma unroll
        for (int v = 0; v < 8; ++v) {
            const int r = rbase + v;
            if (r < nNodes) {
                float val = acc[t][v] + bv;
                if (doRelu) val = fmaxf(val, 0.0f);
                out[(size_t)r * HOUT + col] = val;
            }
        }
    }
}

// ---------------------------------------------------------------------------
// Host orchestration (graph-capture safe: only stream ops)
// ---------------------------------------------------------------------------
extern "C" void kernel_launch(void* const* d_in, const int* in_sizes, int n_in,
                              void* d_out, int out_size, void* d_ws, size_t ws_size,
                              hipStream_t stream)
{
    (void)n_in; (void)out_size; (void)ws_size;

    const float* x    = (const float*)d_in[0];
    const int*   ei   = (const int*)d_in[1];     // [2, E]
    const float* eatt = (const float*)d_in[2];
    const float* mW1  = (const float*)d_in[3];
    const float* mb1  = (const float*)d_in[4];
    const float* mW2  = (const float*)d_in[5];
    const float* mb2  = (const float*)d_in[6];
    const float* Wf   = (const float*)d_in[7];   // [4,16,64] -> flat [64,64]
    const float* bf   = (const float*)d_in[8];
    const float* Wm   = (const float*)d_in[9];   // [2,4,64,64] -> 2x [256,64]
    const float* bm   = (const float*)d_in[10];  // [2,64]
    const float* Wl   = (const float*)d_in[11];  // [4,64,8] -> flat [256,8]
    const float* bl   = (const float*)d_in[12];
    float* out = (float*)d_out;

    const int NF = 16, H = 64, OUT = 8, KP = 4;
    const int nN = in_sizes[0] / NF;
    const int nE = in_sizes[2] / 8;
    const int* src = ei;
    const int* dst = ei + nE;

    // workspace carve-up (f32 units; all offsets 32B-aligned)
    float*    ew   = (float*)d_ws;
    float*    deg  = ew + nE;
    float*    dinv = deg + nN;
    float*    nrm  = dinv + nN;
    float*    stk  = nrm + nE;                        // [nN][256] max
    float*    act  = stk + (size_t)nN * (KP * H);     // [nN][64]
    _Float16* frag = (_Float16*)(act + (size_t)nN * H); // <= 16384 halves

    // --- edge weights + gcn norm ---
    hipMemsetAsync(deg, 0, sizeof(float) * nN, stream);
    edge_mlp_kernel<<<(nE + TPB - 1) / TPB, TPB, 0, stream>>>(
        eatt, mW1, mb1, mW2, mb2, dst, ew, deg, nE);
    dinv_kernel<<<(nN + TPB - 1) / TPB, TPB, 0, stream>>>(deg, dinv, nN);
    norm_kernel<<<(nE + TPB - 1) / TPB, TPB, 0, stream>>>(ew, src, dst, dinv, nrm, nE);

    const int gemmBlocks = (((nN + 15) / 16) + 7) / 8;   // 8 waves / block

    // ---- build stacked features [n][Ktot], Ktot = KP*Hin ----
    auto build_stack = [&](const float* in, int Hin) {
        const int Ktot = KP * Hin;
        hipMemsetAsync(stk, 0, (size_t)nN * Ktot * sizeof(float), stream);
        const long total = (long)nN * Hin;
        const int  logH  = (Hin == 64) ? 6 : 4;
        copy_cols_kernel<<<(int)((total + TPB - 1) / TPB), TPB, 0, stream>>>(
            in, stk, total, Hin - 1, logH, Ktot);
        const long waves  = (Hin == 64) ? (long)nE : ((long)nE + 1) / 2;
        const int  pblks  = (int)((waves * 32 + TPB - 1) / TPB);
        for (int k = 1; k < KP; ++k) {
            const float* pin  = stk + (size_t)(k - 1) * Hin;
            float*       pout = stk + (size_t)k * Hin;
            if (Hin == 64)
                propagate_w64<<<pblks, TPB, 0, stream>>>(pin, pout, src, dst,
                                                         nrm, nE, Ktot);
            else
                propagate_w16<<<pblks, TPB, 0, stream>>>(pin, pout, src, dst,
                                                         nrm, nE, Ktot);
        }
    };

    auto pack = [&](const float* W, int Ktot, int Hout) {
        const int NT = (Hout + 15) / 16;
        const int nfrag = (Ktot / 32) * NT * 32;
        pack_weights_kernel<<<(nfrag + TPB - 1) / TPB, TPB, 0, stream>>>(
            W, frag, Ktot, Hout, NT);
    };

    // Layer 1: [N,16] -> stack [N,64] -> act [N,64]
    build_stack(x, NF);
    pack(Wf, 64, H);
    gemm_wmma<64, 64><<<gemmBlocks, TPB, 0, stream>>>(stk, frag, bf, act, nN, 1);

    // Layer 2: [N,64] -> stack [N,256] -> act
    build_stack(act, H);
    pack(Wm, 256, H);
    gemm_wmma<256, 64><<<gemmBlocks, TPB, 0, stream>>>(stk, frag, bm, act, nN, 1);

    // Layer 3
    build_stack(act, H);
    pack(Wm + (size_t)KP * H * H, 256, H);
    gemm_wmma<256, 64><<<gemmBlocks, TPB, 0, stream>>>(stk, frag, bm + H, act, nN, 1);

    // Layer 4: [N,64] -> stack [N,256] -> out [N,8]
    build_stack(act, H);
    pack(Wl, 256, OUT);
    gemm_wmma<256, 8><<<gemmBlocks, TPB, 0, stream>>>(stk, frag, bl, out, nN, 0);
}
